// DiverseBeamSearchDecoder_56573309223485
// MI455X (gfx1250) — matrix operand order repaired and checked by reading, not compiled, not run on hardware
//
#include <hip/hip_runtime.h>
#include <hip/hip_bf16.h>

// ---------------------------------------------------------------------------
// Diverse beam search decoder for MI455X (gfx1250).
// Dominant cost: per-step logits GEMM [32 x 256] x [256 x 50000], done with
// V_WMMA_F32_16X16X32_BF16 (bf16 operands, f32 accumulate). Wout is packed
// once per launch into the ISA per-lane B-fragment layout; each wave computes
// BOTH 16-row M-tiles so every B fragment load feeds two WMMAs.
// Post-GEMM: single fused pass (per-row chunked max/sumexp + top-4 by
// penalty-adjusted logit), then a tiny logsumexp-merge + global top-4 kernel.
// ---------------------------------------------------------------------------

#define VOCAB 50000
#define DMODEL 256
#define NBEAMS 4
#define MAXLEN 24
#define NGROUPS 2
#define BATCH 8
#define ROWS (BATCH * NBEAMS)        // 32
#define NTILES (VOCAB / 16)          // 3125
#define KTILES (DMODEL / 32)         // 8
#define BMP_WORDS 1568               // ceil(50000/32)=1563, padded
#define CHUNKS 16
#define CHUNK_V (VOCAB / CHUNKS)     // 3125

typedef __attribute__((ext_vector_type(16))) __bf16 v16bf;
typedef __attribute__((ext_vector_type(8)))  float  v8f;

union AFrag { v16bf v; unsigned int u[8]; };
union BF2   { unsigned int u; __bf16 h[2]; };

__device__ __forceinline__ unsigned int pack_bf2(float a, float b) {
  BF2 p; p.h[0] = (__bf16)a; p.h[1] = (__bf16)b; return p.u;
}

__device__ __forceinline__ bool dbs_better(float v1, int i1, float v2, int i2) {
  return (v1 > v2) || (v1 == v2 && i1 < i2);  // jax top_k tie-break: lowest index
}

// Merge per-thread sorted-4 lists held in LDS down to a single top-4 at slot 0.
__device__ __forceinline__ void dbs_topk_reduce(float* sv, int* si, int tid) {
  for (int s = 128; s > 0; s >>= 1) {
    if (tid < s) {
      float a0[4], o0[4], m0[4]; int a1[4], o1[4], m1[4];
#pragma unroll
      for (int q = 0; q < 4; ++q) {
        a0[q] = sv[tid * 4 + q];       a1[q] = si[tid * 4 + q];
        o0[q] = sv[(tid + s) * 4 + q]; o1[q] = si[(tid + s) * 4 + q];
      }
      int ia = 0, ib = 0;
#pragma unroll
      for (int q = 0; q < 4; ++q) {
        if (dbs_better(a0[ia], a1[ia], o0[ib], o1[ib])) { m0[q] = a0[ia]; m1[q] = a1[ia]; ++ia; }
        else                                            { m0[q] = o0[ib]; m1[q] = o1[ib]; ++ib; }
      }
#pragma unroll
      for (int q = 0; q < 4; ++q) { sv[tid * 4 + q] = m0[q]; si[tid * 4 + q] = m1[q]; }
    }
    __syncthreads();
  }
}

__device__ __forceinline__ void dbs_local_insert(float (&bv)[4], int (&bi)[4],
                                                 float cand, int idx) {
  if (dbs_better(cand, idx, bv[3], bi[3])) {
    int p = 3;
    while (p > 0 && dbs_better(cand, idx, bv[p - 1], bi[p - 1])) {
      bv[p] = bv[p - 1]; bi[p] = bi[p - 1]; --p;
    }
    bv[p] = cand; bi[p] = idx;
  }
}

// ------------------------- encoder: ctx = mean_s(src) @ We -----------------
__global__ void dbs_encoder_ctx(const float* __restrict__ src,
                                const float* __restrict__ We,
                                float* __restrict__ ctx) {
  int b = blockIdx.x;          // 8
  int d = threadIdx.x;         // 256
  __shared__ float md[DMODEL];
  float s = 0.f;
  for (int t = 0; t < 128; ++t) s += src[(b * 128 + t) * DMODEL + d];
  md[d] = s * (1.f / 128.f);
  __syncthreads();
  float acc = 0.f;
  for (int dd = 0; dd < DMODEL; ++dd) acc += md[dd] * We[dd * DMODEL + d];
  ctx[b * DMODEL + d] = acc;
}

// ------------------------- init sequences / scores / bitmap ----------------
__global__ void dbs_init_state(int* __restrict__ seqs, float* __restrict__ gs,
                               unsigned int* __restrict__ bmp) {
  int i = blockIdx.x * blockDim.x + threadIdx.x;
  if (i < NGROUPS * ROWS * MAXLEN) seqs[i] = ((i % MAXLEN) == 0) ? 1 : 0; // BOS=1
  if (i < NGROUPS * ROWS)          gs[i]   = ((i & 3) == 0) ? 0.f : -1e9f;
  if (i < NBEAMS * BMP_WORDS)      bmp[i]  = 0u;
}

// ------------------------- pack Wout into WMMA B fragments -----------------
// Bp[(ntile*KTILES + kt)*32 + lane] = 8 dwords (16 bf16) for that lane.
// lane<16 : N = ntile*16+lane,    K = kt*32 + (j<4 ? 2j : 8+2j) + {0,1}
// lane>=16: N = ntile*16+lane-16, K = kt*32 + 8 + (j<4 ? 2j : 8+2j) + {0,1}
__global__ void dbs_pack_wout(const float* __restrict__ Wout,
                              unsigned int* __restrict__ Bp) {
  int tile = blockIdx.x;                 // 0 .. NTILES*KTILES-1
  int ntile = tile >> 3, kt = tile & 7;
  int lane = threadIdx.x & 31;
  int j = threadIdx.x >> 5;              // 0..7
  int n = ntile * 16 + (lane & 15);
  int k = kt * 32 + ((lane < 16) ? 0 : 8) + ((j < 4) ? 2 * j : 8 + 2 * j);
  float f0 = Wout[(size_t)k * VOCAB + n];
  float f1 = Wout[(size_t)(k + 1) * VOCAB + n];
  Bp[((size_t)(ntile * KTILES + kt) * 32 + lane) * 8 + j] = pack_bf2(f0, f1);
}

// ------------------------- h = Emb[tok] + ctx, to bf16 (+ bitmap update) ---
__global__ void dbs_build_h(const float* __restrict__ Emb,
                            const float* __restrict__ ctx,
                            const int* __restrict__ seqs_g,
                            const int* __restrict__ seqs_g0,
                            unsigned int* __restrict__ hbf,
                            unsigned int* __restrict__ bmp,
                            int t, int use_pen) {
  int row = blockIdx.x;                  // 0..31 ; b=row>>2
  int b = row >> 2;
  // fold diversity-bitmap update (group-0 tokens @ t-1) into block 0
  if (use_pen && row == 0 && threadIdx.x < ROWS) {
    int tok0 = seqs_g0[threadIdx.x * MAXLEN + (t - 1)];
    atomicOr(&bmp[(threadIdx.x & 3) * BMP_WORDS + (tok0 >> 5)], 1u << (tok0 & 31));
  }
  int tok = seqs_g[row * MAXLEN + (t - 1)];
  int tid = threadIdx.x;                 // 128, 2 elems each
  float f0 = Emb[(size_t)tok * DMODEL + 2 * tid]     + ctx[b * DMODEL + 2 * tid];
  float f1 = Emb[(size_t)tok * DMODEL + 2 * tid + 1] + ctx[b * DMODEL + 2 * tid + 1];
  hbf[row * (DMODEL / 2) + tid] = pack_bf2(f0, f1);
}

// ------------------------- logits GEMM (WMMA bf16) -------------------------
// One wave per 16-column tile; each wave computes BOTH 16-row M-tiles so each
// B fragment feeds two v_wmma_f32_16x16x32_bf16 ops.
__global__ void dbs_gemm_logits(const unsigned int* __restrict__ Bp,
                                const unsigned int* __restrict__ hbf,
                                float* __restrict__ logits) {
  int wave = threadIdx.x >> 5;
  int lane = threadIdx.x & 31;
  int ntile = blockIdx.x * 8 + wave;
  if (ntile >= NTILES) return;           // uniform per wave -> EXEC all-1s

  v8f c0 = {}, c1 = {};
  const unsigned int* hrow0 = hbf + (lane & 15) * (DMODEL / 2);        // M 0..15
  const unsigned int* hrow1 = hrow0 + 16 * (DMODEL / 2);               // M 16..31
  int base0 = (lane < 16) ? 0 : 4;       // dword offset (K=0 or K=8)
#pragma unroll
  for (int kt = 0; kt < KTILES; ++kt) {
    AFrag a0, a1;
    int b0 = kt * 16 + base0;
#pragma unroll
    for (int j = 0; j < 4; ++j) {
      a0.u[j]     = hrow0[b0 + j];       // K = {0..7} / {8..15}
      a0.u[4 + j] = hrow0[b0 + 8 + j];   // K = {16..23} / {24..31}
      a1.u[j]     = hrow1[b0 + j];
      a1.u[4 + j] = hrow1[b0 + 8 + j];
    }
    const v16bf bf = *(const v16bf*)(Bp + ((size_t)(ntile * KTILES + kt) * 32 + lane) * 8);
    c0 = __builtin_amdgcn_wmma_f32_16x16x32_bf16(false, a0.v, false, bf,
                                                 (short)0, c0, false, false);
    c1 = __builtin_amdgcn_wmma_f32_16x16x32_bf16(false, a1.v, false, bf,
                                                 (short)0, c1, false, false);
  }
  int col   = ntile * 16 + (lane & 15);
  int rbase = (lane < 16) ? 0 : 8;
#pragma unroll
  for (int r = 0; r < 8; ++r) {
    logits[(size_t)(rbase + r) * VOCAB + col]      = c0[r];
    logits[(size_t)(rbase + r + 16) * VOCAB + col] = c1[r];
  }
}

// ---------------- fused chunked row-stats + partial top-4 ------------------
// Grid: 32 rows x 16 chunks. Within a row the log-softmax shift is constant,
// so ranking by y = logit - 0.5*pen(v) equals the final within-row ranking.
__global__ void dbs_scan_chunks(const float* __restrict__ logits,
                                const unsigned int* __restrict__ bmp,
                                float* __restrict__ cmax,   // [32][16]
                                float* __restrict__ csum,   // [32][16]
                                float* __restrict__ pval,   // [32][16][4]
                                int* __restrict__ pidx,     // [32][16][4]
                                int use_pen) {
  int r  = blockIdx.x >> 4;              // row 0..31
  int ch = blockIdx.x & 15;              // chunk
  int k  = r & 3;
  int tid = threadIdx.x;                 // 256
  const float* x = logits + (size_t)r * VOCAB + ch * CHUNK_V;
  const unsigned int* brow = bmp + k * BMP_WORDS;
  int vbase = ch * CHUNK_V;

  __shared__ float red[256];
  __shared__ float sv[256 * 4];
  __shared__ int   si[256 * 4];

  // pass 1: chunk max of raw logits (for logsumexp)
  float m = -__builtin_inff();
  for (int i = tid; i < CHUNK_V; i += 256) m = fmaxf(m, x[i]);
  red[tid] = m; __syncthreads();
  for (int s = 128; s > 0; s >>= 1) {
    if (tid < s) red[tid] = fmaxf(red[tid], red[tid + s]);
    __syncthreads();
  }
  m = red[0]; __syncthreads();

  // pass 2: chunk sumexp + local top-4 by y = logit - 0.5*pen(v)
  float bv[4] = {-__builtin_inff(), -__builtin_inff(), -__builtin_inff(), -__builtin_inff()};
  int   bi[4] = {0x7fffffff, 0x7fffffff, 0x7fffffff, 0x7fffffff};
  float sum = 0.f;
  for (int i = tid; i < CHUNK_V; i += 256) {
    float xi = x[i];
    sum += expf(xi - m);
    int v = vbase + i;
    float y = xi;
    if (use_pen && ((brow[v >> 5] >> (v & 31)) & 1u)) y -= 0.5f;
    dbs_local_insert(bv, bi, y, v);
  }
  red[tid] = sum;
#pragma unroll
  for (int q = 0; q < 4; ++q) { sv[tid * 4 + q] = bv[q]; si[tid * 4 + q] = bi[q]; }
  __syncthreads();
  for (int s = 128; s > 0; s >>= 1) {
    if (tid < s) red[tid] += red[tid + s];
    __syncthreads();
  }
  dbs_topk_reduce(sv, si, tid);
  if (tid == 0) { cmax[r * CHUNKS + ch] = m; csum[r * CHUNKS + ch] = red[0]; }
  if (tid < 4) {
    pval[(r * CHUNKS + ch) * 4 + tid] = sv[tid];
    pidx[(r * CHUNKS + ch) * 4 + tid] = si[tid];
  }
}

// ---------------- logsumexp-merge + global top-4 + beam update -------------
__global__ void dbs_merge_select(const float* __restrict__ cmax,
                                 const float* __restrict__ csum,
                                 const float* __restrict__ pval,
                                 const int* __restrict__ pidx,
                                 float* __restrict__ gs_g,
                                 int* __restrict__ seqs_g, int t) {
  int b = blockIdx.x;                    // 8
  int tid = threadIdx.x;                 // 256
  __shared__ int   sseq[NBEAMS * MAXLEN];
  __shared__ float sv[256 * 4];
  __shared__ int   si[256 * 4];
  __shared__ float sgs[4], slz[4], selv[4];
  __shared__ int   selb[4], selt[4];

  if (tid < NBEAMS * MAXLEN) sseq[tid] = seqs_g[b * NBEAMS * MAXLEN + tid];
  if (tid < 4) {
    int r = b * NBEAMS + tid;
    float M = -__builtin_inff();
    for (int c = 0; c < CHUNKS; ++c) M = fmaxf(M, cmax[r * CHUNKS + c]);
    float S = 0.f;
    for (int c = 0; c < CHUNKS; ++c)
      S += csum[r * CHUNKS + c] * expf(cmax[r * CHUNKS + c] - M);
    slz[tid] = M + logf(S);
    sgs[tid] = gs_g[b * 4 + tid];
  }
  __syncthreads();

  // 256 candidate entries: (k, chunk, q)
  {
    int k = tid >> 6, c = (tid >> 2) & 15, q = tid & 3;
    int r = b * NBEAMS + k;
    float y = pval[(r * CHUNKS + c) * 4 + q];
    int   v = pidx[(r * CHUNKS + c) * 4 + q];
    sv[tid * 4 + 0] = sgs[k] + y - slz[k];
    si[tid * 4 + 0] = k * VOCAB + v;
#pragma unroll
    for (int q2 = 1; q2 < 4; ++q2) {
      sv[tid * 4 + q2] = -__builtin_inff();
      si[tid * 4 + q2] = 0x7fffffff;
    }
  }
  __syncthreads();
  dbs_topk_reduce(sv, si, tid);

  if (tid == 0) {
#pragma unroll
    for (int q = 0; q < 4; ++q) {
      int idx = si[q];
      selv[q] = sv[q];
      selb[q] = idx / VOCAB;
      selt[q] = idx - (idx / VOCAB) * VOCAB;
    }
  }
  __syncthreads();
  if (tid < 4) gs_g[b * 4 + tid] = selv[tid];
  if (tid < NBEAMS * MAXLEN) {
    int k = tid / MAXLEN, p = tid % MAXLEN;
    int nv = (p == t) ? selt[k] : sseq[selb[k] * MAXLEN + p];
    seqs_g[b * NBEAMS * MAXLEN + tid] = nv;
  }
}

// ------------------------- write outputs -----------------------------------
__global__ void dbs_finalize(const int* __restrict__ seqs,
                             const float* __restrict__ gs,
                             float* __restrict__ out) {
  int i = blockIdx.x * blockDim.x + threadIdx.x;
  int* oi = (int*)out;
  if (i < BATCH * NGROUPS * NBEAMS * MAXLEN) {           // 1536
    int b = i / (NGROUPS * NBEAMS * MAXLEN);
    int rem = i - b * (NGROUPS * NBEAMS * MAXLEN);
    int gk = rem / MAXLEN, p = rem % MAXLEN;
    int g = gk >> 2, k = gk & 3;
    oi[i] = seqs[g * ROWS * MAXLEN + (b * NBEAMS + k) * MAXLEN + p];
  }
  if (i < BATCH * NGROUPS * NBEAMS) {                    // 64
    int b = i >> 3, gk = i & 7, g = gk >> 2, k = gk & 3;
    out[BATCH * NGROUPS * NBEAMS * MAXLEN + i] = gs[g * ROWS + b * NBEAMS + k];
  }
}

// ---------------------------------------------------------------------------
extern "C" void kernel_launch(void* const* d_in, const int* in_sizes, int n_in,
                              void* d_out, int out_size, void* d_ws, size_t ws_size,
                              hipStream_t stream) {
  const float* src  = (const float*)d_in[0];
  const float* We   = (const float*)d_in[1];
  const float* Emb  = (const float*)d_in[2];
  const float* Wout = (const float*)d_in[3];
  float* out = (float*)d_out;

  char* ws = (char*)d_ws;
  unsigned int* Bp     = (unsigned int*)(ws + 0);                 // 25,600,000 B
  float*        logits = (float*)(ws + 25600000);                 //  6,400,000 B
  unsigned int* hbf    = (unsigned int*)(ws + 32000000);          //     16,384 B
  float*        ctx    = (float*)(ws + 32016384);                 //      8,192 B
  float*        cmax   = (float*)(ws + 32024576);                 //      2,048 B
  float*        csum   = (float*)(ws + 32026624);                 //      2,048 B
  float*        pval   = (float*)(ws + 32028672);                 //      8,192 B
  int*          pidx   = (int*)(ws + 32036864);                   //      8,192 B
  int*          seqs   = (int*)(ws + 32045056);                   //      6,144 B
  float*        gs     = (float*)(ws + 32051200);                 //        256 B
  unsigned int* bmp    = (unsigned int*)(ws + 32051456);          //     25,088 B

  dbs_encoder_ctx<<<BATCH, DMODEL, 0, stream>>>(src, We, ctx);
  dbs_init_state<<<25, 256, 0, stream>>>(seqs, gs, bmp);
  dbs_pack_wout<<<NTILES * KTILES, 256, 0, stream>>>(Wout, Bp);

  const int gemm_blocks = (NTILES + 7) / 8;       // 391
  for (int g = 0; g < NGROUPS; ++g) {
    int*   seqs_g = seqs + g * ROWS * MAXLEN;
    float* gs_g   = gs + g * ROWS;
    int use_pen = (g == 1) ? 1 : 0;
    for (int t = 1; t < MAXLEN; ++t) {
      dbs_build_h<<<ROWS, 128, 0, stream>>>(Emb, ctx, seqs_g, seqs, hbf, bmp,
                                            t, use_pen);
      dbs_gemm_logits<<<gemm_blocks, 256, 0, stream>>>(Bp, hbf, logits);
      dbs_scan_chunks<<<ROWS * CHUNKS, 256, 0, stream>>>(logits, bmp, cmax, csum,
                                                         pval, pidx, use_pen);
      dbs_merge_select<<<BATCH, 256, 0, stream>>>(cmax, csum, pval, pidx,
                                                  gs_g, seqs_g, t);
    }
  }
  dbs_finalize<<<7, 256, 0, stream>>>(seqs, gs, out);
}
